// SimpleRNN_6734508720671
// MI455X (gfx1250) — compile-verified, matrix operand order checked by reading, not run on hardware
//
#include <hip/hip_runtime.h>
#include <hip/hip_bf16.h>
#include <stdint.h>

// ---------------------------------------------------------------------------
// Problem constants (match the reference)
// ---------------------------------------------------------------------------
#define INP 2048
#define HID 4096
#define OUTD 2048
#define SEQ 4096

#define NBLK 128         // persistent blocks for the recurrence (all resident)
#define RTHREADS 256     // 8 wave32s per block

typedef __attribute__((ext_vector_type(8)))  __bf16 v8bf;
typedef __attribute__((ext_vector_type(16))) __bf16 v16bf;
typedef __attribute__((ext_vector_type(8)))  float  v8f;

// Async-to-LDS builtin operand types (from hipcc diagnostic: pointee is
// 'int __attribute__((vector_size(16)))', global AS src / LDS AS dst)
typedef int v4i __attribute__((vector_size(16)));
typedef __attribute__((address_space(1))) v4i gv4i;
typedef __attribute__((address_space(3))) v4i lv4i;

#if __has_builtin(__builtin_amdgcn_global_load_async_to_lds_b128)
#define HAVE_ASYNC_LDS 1
#endif

// ---------------------------------------------------------------------------
// fp32 -> bf16 conversion (RNE via clang's fptrunc)
// ---------------------------------------------------------------------------
__device__ __forceinline__ unsigned short f2bf(float f) {
  __bf16 h = (__bf16)f;
  return __builtin_bit_cast(unsigned short, h);
}

__global__ __launch_bounds__(256)
void cvt_f32_to_bf16(const float* __restrict__ in, unsigned short* __restrict__ out, int n) {
  int i = blockIdx.x * 256 + threadIdx.x;
  if (i < n) out[i] = f2bf(in[i]);
}

// ---------------------------------------------------------------------------
// WMMA fragment load: lane L holds row/col (L&15), K chunks at kb=(L>=16)*8:
// [kb..kb+7] and [kb+16..kb+23]  (two contiguous 16B reads)
// ---------------------------------------------------------------------------
__device__ __forceinline__ v16bf load_frag(const __bf16* p) {
  v8bf lo = *(const v8bf*)(p);
  v8bf hi = *(const v8bf*)(p + 16);
  v16bf f;
#pragma unroll
  for (int i = 0; i < 8; ++i) { f[i] = lo[i]; f[i + 8] = hi[i]; }
  return f;
}

__device__ __forceinline__ v8f wmma_bf16(v16bf a, v16bf b, v8f c) {
  return __builtin_amdgcn_wmma_f32_16x16x32_bf16(
      /*neg_a=*/false, a, /*neg_b=*/false, b,
      /*c_mod=*/(short)0, c, /*reuse_a=*/false, /*reuse_b=*/false);
}

// ---------------------------------------------------------------------------
// bf16 WMMA GEMM:  C[M,N](fp32) = A[M,K](bf16 rm) @ B[N,K](bf16 rm)^T + bias[N]
// 2x2 register blocking: each wave owns a 32x32 macro-tile (4 WMMAs / K-step,
// 2 A-frags + 2 B-frags reused => 16 FLOP per global byte).
// Block = 8 waves = 64(M) x 128(N); K-loop steps of 32.
// ---------------------------------------------------------------------------
__global__ __launch_bounds__(256)
void gemm_bf16_wmma(const unsigned short* __restrict__ Au,
                    const unsigned short* __restrict__ Bu,
                    const float* __restrict__ bias,
                    float* __restrict__ C,
                    int K, int lda, int ldb, int ldc) {
  const __bf16* A = reinterpret_cast<const __bf16*>(Au);
  const __bf16* B = reinterpret_cast<const __bf16*>(Bu);

  const int lane = threadIdx.x & 31;
  const int wave = threadIdx.x >> 5;

  const int tm = blockIdx.y * 64  + (wave >> 2) * 32;  // macro-tile row (M)
  const int tn = blockIdx.x * 128 + (wave & 3) * 32;   // macro-tile col (N)

  const int l15 = lane & 15;
  const int kb  = (lane >> 4) << 3;                    // 0 or 8

  const __bf16* a0p = A + (size_t)(tm + l15) * lda + kb;
  const __bf16* a1p = a0p + (size_t)16 * lda;
  const __bf16* b0p = B + (size_t)(tn + l15) * ldb + kb;
  const __bf16* b1p = b0p + (size_t)16 * ldb;

  v8f c00 = {}, c01 = {}, c10 = {}, c11 = {};

  for (int k0 = 0; k0 < K; k0 += 32) {
    v16bf a0 = load_frag(a0p);
    v16bf a1 = load_frag(a1p);
    v16bf b0 = load_frag(b0p);
    v16bf b1 = load_frag(b1p);

    // Prefetch ~4 K-steps ahead on all four streams (speculative)
    __builtin_prefetch(a0p + 128, 0, 1);
    __builtin_prefetch(a1p + 128, 0, 1);
    __builtin_prefetch(b0p + 128, 0, 1);
    __builtin_prefetch(b1p + 128, 0, 1);

    c00 = wmma_bf16(a0, b0, c00);
    c01 = wmma_bf16(a0, b1, c01);
    c10 = wmma_bf16(a1, b0, c10);
    c11 = wmma_bf16(a1, b1, c11);

    a0p += 32; a1p += 32; b0p += 32; b1p += 32;
  }

  const float bv0 = bias ? bias[tn + l15] : 0.0f;
  const float bv1 = bias ? bias[tn + 16 + l15] : 0.0f;
  // C/D layout: VGPR r -> row (tile_m + r + (lane>=16)*8), col tile_n + (lane&15)
  const int r0 = tm + kb;
#pragma unroll
  for (int r = 0; r < 8; ++r) {
    C[(size_t)(r0 + r)      * ldc + tn + l15]      = c00[r] + bv0;
    C[(size_t)(r0 + r)      * ldc + tn + 16 + l15] = c01[r] + bv1;
    C[(size_t)(r0 + 16 + r) * ldc + tn + l15]      = c10[r] + bv0;
    C[(size_t)(r0 + 16 + r) * ldc + tn + 16 + l15] = c11[r] + bv1;
  }
}

// ---------------------------------------------------------------------------
// Barrier / state init (runs before the persistent kernel, stream-ordered)
// ---------------------------------------------------------------------------
__global__ void init_state(float* __restrict__ hbuf, int* __restrict__ syncb) {
  for (int j = threadIdx.x; j < HID; j += 256) hbuf[j] = 0.0f;  // h0 = 0
  if (threadIdx.x < 2) syncb[threadIdx.x * 64] = 0;             // count, generation
}

// ---------------------------------------------------------------------------
// Device-wide generation barrier (persistent kernel, NBLK blocks resident)
// ---------------------------------------------------------------------------
__device__ __forceinline__ void grid_sync(int* cnt, int* gen, int nb) {
  __syncthreads();
  __threadfence();
  if (threadIdx.x == 0) {
    int g = __hip_atomic_load(gen, __ATOMIC_RELAXED, __HIP_MEMORY_SCOPE_AGENT);
    if (__hip_atomic_fetch_add(cnt, 1, __ATOMIC_ACQ_REL, __HIP_MEMORY_SCOPE_AGENT) == nb - 1) {
      __hip_atomic_store(cnt, 0, __ATOMIC_RELAXED, __HIP_MEMORY_SCOPE_AGENT);
      __hip_atomic_fetch_add(gen, 1, __ATOMIC_ACQ_REL, __HIP_MEMORY_SCOPE_AGENT);
    } else {
      while (__hip_atomic_load(gen, __ATOMIC_ACQUIRE, __HIP_MEMORY_SCOPE_AGENT) == g) {
        __builtin_amdgcn_s_sleep(2);
      }
    }
  }
  __syncthreads();
}

// ---------------------------------------------------------------------------
// Sequential recurrence: h' = tanh(x_proj[t] + W_hh(bf16) @ h + b_hh)
// W_hh bf16 = 32 MB -> L2-resident; matvec is bandwidth-bound (1 FLOP/byte),
// so VALU dot + wave32 shfl reduction. 128 persistent blocks pull aggregate
// L2 bandwidth; h staged into LDS each step via async-to-LDS DMA (ASYNCcnt).
// ---------------------------------------------------------------------------
__global__ __launch_bounds__(RTHREADS)
void rnn_recurrence(const unsigned short* __restrict__ Whhu,
                    const float* __restrict__ xproj,
                    const float* __restrict__ bhh,
                    float* __restrict__ hbuf,            // 2 * HID fp32 (ping-pong)
                    unsigned short* __restrict__ hall,   // SEQ * HID bf16
                    int* __restrict__ syncb) {
  const __bf16* Whh = reinterpret_cast<const __bf16*>(Whhu);
  __shared__ float h_s[HID];                             // 16 KB

  const int lane = threadIdx.x & 31;
  const int wave = threadIdx.x >> 5;
  int* cnt = syncb;
  int* gen = syncb + 64;                                 // separate cacheline

  for (int t = 0; t < SEQ; ++t) {
    const float* hcur = hbuf + (size_t)(t & 1) * HID;
    float*       hnxt = hbuf + (size_t)((t + 1) & 1) * HID;

#ifdef HAVE_ASYNC_LDS
    {
      // 16 KB -> 1024 async b128 transfers, 4 per thread (ASYNCcnt-tracked)
      for (int i = threadIdx.x; i < HID / 4; i += RTHREADS) {
        __builtin_amdgcn_global_load_async_to_lds_b128(
            (gv4i*)(hcur + i * 4),
            (lv4i*)(h_s + i * 4),
            0, 0);
      }
#if __has_builtin(__builtin_amdgcn_s_wait_asynccnt)
      __builtin_amdgcn_s_wait_asynccnt(0);
#else
      asm volatile("s_wait_asynccnt 0x0" ::: "memory");
#endif
    }
#else
    for (int i = threadIdx.x; i < HID; i += RTHREADS) h_s[i] = hcur[i];
#endif
    __syncthreads();

    const float* xp = xproj + (size_t)t * HID;

#pragma unroll
    for (int rr = 0; rr < HID / NBLK / 8; ++rr) {        // 4 rows per wave
      const int j = blockIdx.x * (HID / NBLK) + wave * (HID / NBLK / 8) + rr;
      const v8bf* wrow = (const v8bf*)(Whh + (size_t)j * HID);
      float acc = 0.0f;
#pragma unroll
      for (int i = 0; i < HID / (8 * 32); ++i) {         // 16 chunks of 8 bf16
        const int c = i * 32 + lane;
        v8bf w = wrow[c];
        const float* hp = &h_s[c * 8];
#pragma unroll
        for (int e = 0; e < 8; ++e) acc += (float)w[e] * hp[e];
      }
#pragma unroll
      for (int off = 16; off >= 1; off >>= 1)
        acc += __shfl_xor(acc, off, 32);
      if (lane == 0) {
        float v = tanhf(acc + xp[j] + bhh[j]);
        hnxt[j] = v;
        hall[(size_t)t * HID + j] = f2bf(v);
      }
    }
    grid_sync(cnt, gen, NBLK);
  }
}

// ---------------------------------------------------------------------------
// Launch: convert -> GEMM1 (x_proj) -> init -> recurrence -> GEMM2 (logits)
// ---------------------------------------------------------------------------
extern "C" void kernel_launch(void* const* d_in, const int* in_sizes, int n_in,
                              void* d_out, int out_size, void* d_ws, size_t ws_size,
                              hipStream_t stream) {
  const float* x_seq = (const float*)d_in[0];  // [SEQ, INP]
  const float* W_ih  = (const float*)d_in[1];  // [HID, INP]
  const float* b_ih  = (const float*)d_in[2];  // [HID]
  const float* W_hh  = (const float*)d_in[3];  // [HID, HID]
  const float* b_hh  = (const float*)d_in[4];  // [HID]
  const float* W_out = (const float*)d_in[5];  // [OUTD, HID]
  const float* b_out = (const float*)d_in[6];  // [OUTD]
  float* logits = (float*)d_out;               // [SEQ, OUTD]

  // Workspace layout (all offsets 256B-aligned), total ~185 MB
  char* ws = (char*)d_ws;
  constexpr size_t SZ_XBF   = (size_t)SEQ * INP * 2;   // 16 MiB
  constexpr size_t SZ_WIH   = (size_t)HID * INP * 2;   // 16 MiB
  constexpr size_t SZ_WHH   = (size_t)HID * HID * 2;   // 32 MiB
  constexpr size_t SZ_WOUT  = (size_t)OUTD * HID * 2;  // 16 MiB
  constexpr size_t SZ_XPROJ = (size_t)SEQ * HID * 4;   // 64 MiB
  constexpr size_t SZ_HALL  = (size_t)SEQ * HID * 2;   // 32 MiB

  unsigned short* x_bf    = (unsigned short*)(ws);
  unsigned short* wih_bf  = (unsigned short*)(ws + SZ_XBF);
  unsigned short* whh_bf  = (unsigned short*)(ws + SZ_XBF + SZ_WIH);
  unsigned short* wout_bf = (unsigned short*)(ws + SZ_XBF + SZ_WIH + SZ_WHH);
  float*          xproj   = (float*)(ws + SZ_XBF + SZ_WIH + SZ_WHH + SZ_WOUT);
  unsigned short* hall    = (unsigned short*)(ws + SZ_XBF + SZ_WIH + SZ_WHH + SZ_WOUT + SZ_XPROJ);
  float*          hbuf    = (float*)(ws + SZ_XBF + SZ_WIH + SZ_WHH + SZ_WOUT + SZ_XPROJ + SZ_HALL);
  int*            syncb   = (int*)(ws + SZ_XBF + SZ_WIH + SZ_WHH + SZ_WOUT + SZ_XPROJ + SZ_HALL
                                      + (size_t)2 * HID * 4);

  // 1) fp32 -> bf16 conversions
  {
    int n;
    n = SEQ * INP;  cvt_f32_to_bf16<<<(n + 255) / 256, 256, 0, stream>>>(x_seq, x_bf, n);
    n = HID * INP;  cvt_f32_to_bf16<<<(n + 255) / 256, 256, 0, stream>>>(W_ih, wih_bf, n);
    n = HID * HID;  cvt_f32_to_bf16<<<(n + 255) / 256, 256, 0, stream>>>(W_hh, whh_bf, n);
    n = OUTD * HID; cvt_f32_to_bf16<<<(n + 255) / 256, 256, 0, stream>>>(W_out, wout_bf, n);
  }

  // 2) GEMM1: x_proj[SEQ,HID] = x_bf @ wih_bf^T + b_ih   (K = INP)
  {
    dim3 grid(HID / 128, SEQ / 64);
    gemm_bf16_wmma<<<grid, 256, 0, stream>>>(x_bf, wih_bf, b_ih, xproj,
                                             INP, INP, INP, HID);
  }

  // 3) init barrier + h0
  init_state<<<1, 256, 0, stream>>>(hbuf, syncb);

  // 4) persistent recurrence over SEQ steps
  rnn_recurrence<<<NBLK, RTHREADS, 0, stream>>>(whh_bf, xproj, b_hh, hbuf, hall, syncb);

  // 5) GEMM2: logits[SEQ,OUTD] = hall @ wout_bf^T + b_out   (K = HID)
  {
    dim3 grid(OUTD / 128, SEQ / 64);
    gemm_bf16_wmma<<<grid, 256, 0, stream>>>(hall, wout_bf, b_out, logits,
                                             HID, HID, HID, OUTD);
  }
}
